// ResidualQuantizer_19542101197039
// MI455X (gfx1250) — compile-verified, matrix operand order checked by reading, not run on hardware
//
#include <hip/hip_runtime.h>

#define NUM_LAYERS 3
#define K_CODES 4096
#define DIM 256
#define N_TOKENS 65536

typedef __attribute__((ext_vector_type(16))) __bf16 v16bf;
typedef __attribute__((ext_vector_type(8)))  float  v8f;
typedef __attribute__((ext_vector_type(8)))  unsigned short u16x8;
typedef __attribute__((ext_vector_type(16))) unsigned short u16x16;

__device__ __forceinline__ unsigned short f32_to_bf16_rne(float f) {
  union { float f; unsigned int u; } c; c.f = f;
  unsigned int u = c.u;
  u += 0x7FFFu + ((u >> 16) & 1u);   // round-to-nearest-even
  return (unsigned short)(u >> 16);
}

// One block per codebook row (3*4096 rows): write bf16 copy + ||c||^2.
__global__ void __launch_bounds__(256)
cb_prep_kernel(const float* __restrict__ codebooks,
               unsigned short* __restrict__ cb_bf16,
               float* __restrict__ cnorm) {
  __shared__ float red[256];
  const int row = blockIdx.x;
  const int t = threadIdx.x;
  float v = codebooks[(size_t)row * DIM + t];
  cb_bf16[(size_t)row * DIM + t] = f32_to_bf16_rne(v);
  red[t] = v * v;
  __syncthreads();
  #pragma unroll
  for (int s = 128; s > 0; s >>= 1) {
    if (t < s) red[t] += red[t + s];
    __syncthreads();
  }
  if (t == 0) cnorm[row] = red[0];
}

// r = z - q (or r = z and q = 0 for layer 0); write f32 residual + bf16 copy.
__global__ void __launch_bounds__(256)
residual_kernel(const float* __restrict__ z,
                float* __restrict__ q,            // running quantized (zero-init if layer0)
                float* __restrict__ resid_out,    // d_out residual slab for this layer
                unsigned short* __restrict__ r_bf16,
                int layer0) {
  const size_t i = ((size_t)blockIdx.x * blockDim.x + threadIdx.x) * 4;
  const float4 zv = *(const float4*)(z + i);
  float4 r;
  if (layer0) {
    r = zv;
    *(float4*)(q + i) = make_float4(0.f, 0.f, 0.f, 0.f);
  } else {
    const float4 qv = *(const float4*)(q + i);
    r = make_float4(zv.x - qv.x, zv.y - qv.y, zv.z - qv.z, zv.w - qv.w);
  }
  *(float4*)(resid_out + i) = r;
  ushort4 rb;
  rb.x = f32_to_bf16_rne(r.x);
  rb.y = f32_to_bf16_rne(r.y);
  rb.z = f32_to_bf16_rne(r.z);
  rb.w = f32_to_bf16_rne(r.w);
  *(ushort4*)(r_bf16 + i) = rb;
}

__device__ __forceinline__ v16bf load_frag_row(const unsigned short* p) {
  // 8 bf16 at p, 8 bf16 at p+16 elements: the {K..K+7, K+16..K+23} fragment halves.
  u16x8 lo = *(const u16x8*)(p);
  u16x8 hi = *(const u16x8*)(p + 16);
  u16x16 raw;
  #pragma unroll
  for (int j = 0; j < 8; ++j) { raw[j] = lo[j]; raw[j + 8] = hi[j]; }
  return __builtin_bit_cast(v16bf, raw);
}

// Fused bf16 WMMA GEMM (codebook x residual^T) + argmin epilogue.
// Block = 4 waves. Waves pair up on a 32-token tile-pair and split-K the
// codebook (parity 0 -> entries [0,2048), parity 1 -> [2048,4096)); each wave
// runs the 16-WMMA inner loop (2 token tiles per A fragment => 1 b128 load per
// WMMA), then the two candidate streams merge through 512 B of LDS.
__global__ void __launch_bounds__(128)
gemm_argmin_kernel(const unsigned short* __restrict__ r_bf16,   // [N_TOKENS][DIM]
                   const unsigned short* __restrict__ cb_bf16,  // [K_CODES][DIM] (layer slab)
                   const float* __restrict__ cnorm,             // [K_CODES]    (layer slab)
                   float* __restrict__ codes_f,                 // [N_TOKENS] in d_out
                   int* __restrict__ codes_i) {                 // [N_TOKENS] scratch
  const int lane  = threadIdx.x & 31;
  const int wave  = threadIdx.x >> 5;
  const int pp    = wave >> 1;               // which tile-pair within the block
  const int par   = wave & 1;                // codebook half
  const int pair  = blockIdx.x * 2 + pp;     // token tile-pair index
  const int tile0 = pair * 2;
  const int tile1 = tile0 + 1;
  const int half  = lane >> 4;               // 0 or 1
  const int l15   = lane & 15;
  const int kbase = half * 8;

  // Preload B fragments for both token tiles (kept in VGPRs for the whole run).
  const unsigned short* rrow0 = r_bf16 + (size_t)(tile0 * 16 + l15) * DIM + kbase;
  const unsigned short* rrow1 = r_bf16 + (size_t)(tile1 * 16 + l15) * DIM + kbase;
  v16bf bfrag0[8], bfrag1[8];
  #pragma unroll
  for (int c = 0; c < 8; ++c) {
    bfrag0[c] = load_frag_row(rrow0 + c * 32);
    bfrag1[c] = load_frag_row(rrow1 + c * 32);
  }

  // 8 independent per-slot running minima per tile (slot j <-> M = half*8 + j).
  float bv0[8], bv1[8];
  int   bi0[8], bi1[8];
  #pragma unroll
  for (int j = 0; j < 8; ++j) {
    bv0[j] = 3.402823466e+38f; bv1[j] = 3.402823466e+38f;
    bi0[j] = 0;                bi1[j] = 0;
  }

  const unsigned short* abase = cb_bf16 + (size_t)l15 * DIM + kbase;
  const int mbeg = par * (K_CODES / 32);     // 128 m-tiles per wave
  const int mend = mbeg + (K_CODES / 32);

  for (int m = mbeg; m < mend; ++m) {
    const unsigned short* arow = abase + (size_t)m * 16 * DIM;
    // Get the next codebook tile moving while this one computes.
    __builtin_prefetch(arow + (size_t)16 * DIM, 0, 1);

    v8f acc0 = {};
    v8f acc1 = {};
    #pragma unroll
    for (int c = 0; c < 8; ++c) {
      v16bf afrag = load_frag_row(arow + c * 32);
      acc0 = __builtin_amdgcn_wmma_f32_16x16x32_bf16(
          false, afrag, false, bfrag0[c], (short)0, acc0, false, false);
      acc1 = __builtin_amdgcn_wmma_f32_16x16x32_bf16(
          false, afrag, false, bfrag1[c], (short)0, acc1, false, false);
    }

    // acc[j] = r_token . c_{m*16 + half*8 + j}; score = ||c||^2 - 2 r.c
    const int base = m * 16 + half * 8;
    const float4 cn0 = *(const float4*)(cnorm + base);
    const float4 cn1 = *(const float4*)(cnorm + base + 4);
    const float cn[8] = {cn0.x, cn0.y, cn0.z, cn0.w, cn1.x, cn1.y, cn1.z, cn1.w};
    #pragma unroll
    for (int j = 0; j < 8; ++j) {
      const float s0 = cn[j] - 2.0f * acc0[j];
      const float s1 = cn[j] - 2.0f * acc1[j];
      if (s0 < bv0[j]) { bv0[j] = s0; bi0[j] = base + j; }
      if (s1 < bv1[j]) { bv1[j] = s1; bi1[j] = base + j; }
    }
  }

  // Merge the 8 slots (tie -> lower index preserves argmin-first semantics).
  float bestv0 = bv0[0], bestv1 = bv1[0];
  int   besti0 = bi0[0], besti1 = bi1[0];
  #pragma unroll
  for (int j = 1; j < 8; ++j) {
    if (bv0[j] < bestv0 || (bv0[j] == bestv0 && bi0[j] < besti0)) { bestv0 = bv0[j]; besti0 = bi0[j]; }
    if (bv1[j] < bestv1 || (bv1[j] == bestv1 && bi1[j] < besti1)) { bestv1 = bv1[j]; besti1 = bi1[j]; }
  }

  // Merge the two half-lane candidate streams (lane L vs L+16).
  {
    const float ov = __shfl_xor(bestv0, 16, 32);
    const int   oi = __shfl_xor(besti0, 16, 32);
    if (ov < bestv0 || (ov == bestv0 && oi < besti0)) { bestv0 = ov; besti0 = oi; }
  }
  {
    const float ov = __shfl_xor(bestv1, 16, 32);
    const int   oi = __shfl_xor(besti1, 16, 32);
    if (ov < bestv1 || (ov == bestv1 && oi < besti1)) { bestv1 = ov; besti1 = oi; }
  }

  // Merge the split-K halves through LDS. Parity-0 indices are always lower,
  // so on ties parity 0 wins (strict < for the parity-1 candidate).
  __shared__ float s_val[2][32];
  __shared__ int   s_idx[2][32];
  if (par == 0 && half == 0) {
    s_val[pp][l15]      = bestv0; s_idx[pp][l15]      = besti0;
    s_val[pp][16 + l15] = bestv1; s_idx[pp][16 + l15] = besti1;
  }
  __syncthreads();
  if (par == 1 && half == 0) {
    float v0 = s_val[pp][l15];      int i0 = s_idx[pp][l15];
    float v1 = s_val[pp][16 + l15]; int i1 = s_idx[pp][16 + l15];
    if (bestv0 < v0) { i0 = besti0; }
    if (bestv1 < v1) { i1 = besti1; }
    const int token0 = tile0 * 16 + l15;
    const int token1 = tile1 * 16 + l15;
    codes_i[token0] = i0;
    codes_f[token0] = (float)i0;
    codes_i[token1] = i1;
    codes_f[token1] = (float)i1;
  }
}

// q[token] += W[code[token]] (f32 codebook), 4 tokens per 256-thread block.
__global__ void __launch_bounds__(256)
gather_add_kernel(const float* __restrict__ cb,      // layer slab [K_CODES][DIM]
                  const int* __restrict__ codes_i,
                  float* __restrict__ q) {
  const int token = blockIdx.x * 4 + (threadIdx.x >> 6);
  const int d = (threadIdx.x & 63) * 4;
  const int code = codes_i[token];
  const float4 w = *(const float4*)(cb + (size_t)code * DIM + d);
  float4* qp = (float4*)(q + (size_t)token * DIM + d);
  float4 qv = *qp;
  qv.x += w.x; qv.y += w.y; qv.z += w.z; qv.w += w.w;
  *qp = qv;
}

extern "C" void kernel_launch(void* const* d_in, const int* in_sizes, int n_in,
                              void* d_out, int out_size, void* d_ws, size_t ws_size,
                              hipStream_t stream) {
  const float* z         = (const float*)d_in[0];
  const float* codebooks = (const float*)d_in[1];

  // d_out layout (flat f32): codes [3*N] | quantized [N*DIM] | residuals [3*N*DIM]
  float* out       = (float*)d_out;
  float* codes_out = out;
  float* quant     = out + (size_t)NUM_LAYERS * N_TOKENS;
  float* resid     = quant + (size_t)N_TOKENS * DIM;

  // Workspace carve-up (~40 MB)
  char* w = (char*)d_ws;
  unsigned short* r_bf16  = (unsigned short*)w;  w += (size_t)N_TOKENS * DIM * 2;
  unsigned short* cb_bf16 = (unsigned short*)w;  w += (size_t)NUM_LAYERS * K_CODES * DIM * 2;
  float*          cnorm   = (float*)w;           w += (size_t)NUM_LAYERS * K_CODES * 4;
  int*            codes_i = (int*)w;             w += (size_t)N_TOKENS * 4;
  (void)in_sizes; (void)n_in; (void)out_size; (void)ws_size;

  // Precompute bf16 codebooks + ||c||^2 for all layers.
  cb_prep_kernel<<<NUM_LAYERS * K_CODES, 256, 0, stream>>>(codebooks, cb_bf16, cnorm);

  for (int layer = 0; layer < NUM_LAYERS; ++layer) {
    residual_kernel<<<(N_TOKENS * DIM / 4) / 256, 256, 0, stream>>>(
        z, quant, resid + (size_t)layer * N_TOKENS * DIM, r_bf16, layer == 0 ? 1 : 0);

    // 2048 tile-pairs, 2 pairs per block, split-K over wave parity -> 4096 waves.
    gemm_argmin_kernel<<<(N_TOKENS / 32) / 2, 128, 0, stream>>>(
        r_bf16,
        cb_bf16 + (size_t)layer * K_CODES * DIM,
        cnorm + (size_t)layer * K_CODES,
        codes_out + (size_t)layer * N_TOKENS,
        codes_i);

    gather_add_kernel<<<N_TOKENS / 4, 256, 0, stream>>>(
        codebooks + (size_t)layer * K_CODES * DIM, codes_i, quant);
  }
}
